// Generator_5299989643940
// MI455X (gfx1250) — compile-verified, matrix operand order checked by reading, not run on hardware
//
#include <hip/hip_runtime.h>
#include <hip/hip_bf16.h>

#define M_NODES 10000
#define R_DIM   10
#define Q_DIM   48
#define NHID    48
#define G4      192     // 4*NH
#define HID0    16
#define NEDGE   640000
#define T_STEPS 10

typedef __attribute__((ext_vector_type(16))) _Float16 v16h;
typedef __attribute__((ext_vector_type(8)))  float    v8f;

// ---------------- elementwise utility kernels ----------------
__global__ void zero_f32(float* __restrict__ p, int n) {
  int i = blockIdx.x * blockDim.x + threadIdx.x;
  if (i < n) p[i] = 0.f;
}

__global__ void copy_f32(const float* __restrict__ s, float* __restrict__ d, int n) {
  int i = blockIdx.x * blockDim.x + threadIdx.x;
  if (i < n) d[i] = s[i];
}

__global__ void deg_accum(const long long* __restrict__ dst, float* __restrict__ deg, int e) {
  int i = blockIdx.x * blockDim.x + threadIdx.x;
  if (i < e) atomicAdd(&deg[(int)dst[i]], 1.0f);
}

__global__ void deg_finalize(const float* __restrict__ deg, float* __restrict__ dinv, int n) {
  int i = blockIdx.x * blockDim.x + threadIdx.x;
  if (i < n) dinv[i] = rsqrtf(deg[i] + 1.0f);   // +1 for self loop
}

// ---------------- WMMA GEMM: Y[M,N] = X[M,K] * B + bias ----------------
// B(k,n) = TRANSB ? Wm[n*K + k] : Wm[k*N + n]
// One wave per 16x16 output tile, Mrows % 16 == 0 (no M bounds checks).
// K padded to a multiple of 32 with clamp-and-mask loads (no divergent
// branches: unconditional in-bounds float2 loads + v_cndmask zeroing).
template <int KDIM, int NDIM, bool TRANSB>
__launch_bounds__(256)
__global__ void wmma_gemm_f16(const float* __restrict__ X,
                              const float* __restrict__ Wm,
                              const float* __restrict__ bias,
                              float* __restrict__ Y, int Mrows) {
  static_assert((KDIM & 1) == 0, "KDIM must be even");
  constexpr int NT = NDIM / 16;
  int wid    = (blockIdx.x * blockDim.x + threadIdx.x) >> 5;
  int mtiles = Mrows >> 4;                        // exact: 10000/16 = 625
  if (wid >= mtiles * NT) return;                 // whole-wave uniform exit
  int mtile = wid / NT;
  int ntile = wid - mtile * NT;
  int lane  = threadIdx.x & 31;
  int half  = lane >> 4;
  int r16   = lane & 15;
  int mrow  = mtile * 16 + r16;                   // A: M = lane%16
  int ncol  = ntile * 16 + r16;                   // B: N = lane%16

  const float* __restrict__ Arow = X + (size_t)mrow * KDIM;
  const float* __restrict__ Bp   = TRANSB ? (Wm + (size_t)ncol * KDIM) : (Wm + ncol);

  v8f acc;
  float bv = bias ? bias[ncol] : 0.f;
#pragma unroll
  for (int v = 0; v < 8; ++v) acc[v] = bv;

#pragma unroll
  for (int k0 = 0; k0 < KDIM; k0 += 32) {
    v16h a, b;
#pragma unroll
    for (int v = 0; v < 8; ++v) {
      // 16-bit A 16x32 layout: K = (v%4)*2 + half*8 + (v/4)*16
      int  kk = k0 + ((v & 3) << 1) + (half << 3) + ((v >> 2) << 4);
      int  kc = (kk <= KDIM - 2) ? kk : (KDIM - 2);   // clamped, stays even
      bool m0 = kk < KDIM;
      bool m1 = kk + 1 < KDIM;
      float2 xa = *reinterpret_cast<const float2*>(Arow + kc);
      a[2 * v]     = (_Float16)(m0 ? xa.x : 0.f);
      a[2 * v + 1] = (_Float16)(m1 ? xa.y : 0.f);
      if (TRANSB) {                                   // contiguous along K
        float2 xb = *reinterpret_cast<const float2*>(Bp + kc);
        b[2 * v]     = (_Float16)(m0 ? xb.x : 0.f);
        b[2 * v + 1] = (_Float16)(m1 ? xb.y : 0.f);
      } else {                                        // strided (tiny weight, L2-hot)
        float b0 = Bp[(size_t)kc * NDIM];
        float b1 = Bp[(size_t)(kc + 1) * NDIM];
        b[2 * v]     = (_Float16)(m0 ? b0 : 0.f);
        b[2 * v + 1] = (_Float16)(m1 ? b1 : 0.f);
      }
    }
    acc = __builtin_amdgcn_wmma_f32_16x16x32_f16(false, a, false, b, (short)0, acc,
                                                 false, false);
  }
#pragma unroll
  for (int v = 0; v < 8; ++v) {                    // C/D: M = v + 8*half, N = lane%16
    int m = mtile * 16 + v + (half << 3);
    Y[(size_t)m * NDIM + ncol] = acc[v];           // unconditional store
  }
}

// ---------------- GCN edge scatter: agg[dst] += xw[src] * dinv[src]*dinv[dst] ----------------
// 2-D grid: x = edges, y = float4 channel group (no integer division).
template <int C>
__global__ void gcn_edge_agg(const float* __restrict__ XW,
                             const long long* __restrict__ src,
                             const long long* __restrict__ dst,
                             const float* __restrict__ dinv,
                             float* __restrict__ agg) {
  int e = blockIdx.x * blockDim.x + threadIdx.x;
  if (e >= NEDGE) return;
  int c = blockIdx.y << 2;
  int s = (int)src[e];
  int d = (int)dst[e];
  float coef = dinv[s] * dinv[d];
  const float4 x = *reinterpret_cast<const float4*>(XW + s * C + c);
  atomicAdd(&agg[d * C + c + 0], x.x * coef);
  atomicAdd(&agg[d * C + c + 1], x.y * coef);
  atomicAdd(&agg[d * C + c + 2], x.z * coef);
  atomicAdd(&agg[d * C + c + 3], x.w * coef);
}

// ---------------- self-loop term + bias (+optional sigmoid) ----------------
template <int C, bool SIG>
__global__ void gcn_self_bias(const float* __restrict__ XW,
                              const float* __restrict__ dinv,
                              const float* __restrict__ bias,
                              const float* __restrict__ agg,
                              float* __restrict__ out, int n) {
  int i = blockIdx.x * blockDim.x + threadIdx.x;
  if (i >= n * C) return;
  int node = i / C;
  int c    = i - node * C;
  float di = dinv[node];
  float v  = agg[i] + XW[i] * di * di + bias[c];
  out[i] = SIG ? (1.f / (1.f + expf(-v))) : v;
}

// ---------------- fused sequential LSTM + dense-tanh accumulation ----------------
// Single block, 192 threads. Thread t owns gate row t (W_hh row in 48 VGPRs).
// h lives in LDS; c lives in registers of threads 0..47. State persists in
// `state` across launches (reference carries h,c through all 20 sequences).
__launch_bounds__(192)
__global__ void lstm_dense_kernel(const float* __restrict__ XG,       // [S,192] x@W_ih.T+b_ih
                                  const float* __restrict__ W_hh,     // [192,48]
                                  const float* __restrict__ b_hh,     // [192]
                                  const float* __restrict__ dense_w,  // [10,48]
                                  const float* __restrict__ dense_b,  // [10]
                                  float* __restrict__ state,          // [96] h|c
                                  float* __restrict__ out,            // [S,10], += tanh(...)
                                  int S) {
  __shared__ float h_lds[NHID];
  __shared__ float g_lds[G4];
  __shared__ float dw[R_DIM * NHID];
  __shared__ float db[R_DIM];
  int t = threadIdx.x;

  float wrow[NHID];
#pragma unroll
  for (int k = 0; k < NHID; ++k) wrow[k] = W_hh[t * NHID + k];
  float bh = b_hh[t];
  float c  = 0.f;
  if (t < NHID) { h_lds[t] = state[t]; c = state[NHID + t]; }
  for (int i = t; i < R_DIM * NHID; i += G4) dw[i] = dense_w[i];
  if (t < R_DIM) db[t] = dense_b[t];
  __syncthreads();

  float xg = XG[t];                                 // prefetch row 0
  for (int s = 0; s < S; ++s) {
    float xg_next = (s + 1 < S) ? XG[(s + 1) * G4 + t] : 0.f;  // overlap latency
    float g = xg + bh;
#pragma unroll
    for (int k = 0; k < NHID; ++k) g += wrow[k] * h_lds[k];
    // gate order i,f,g,o over rows 0..191
    float a = (t >= 2 * NHID && t < 3 * NHID) ? tanhf(g)
                                              : (1.f / (1.f + expf(-g)));
    g_lds[t] = a;
    __syncthreads();
    if (t < NHID) {
      float ig = g_lds[t], fg = g_lds[NHID + t];
      float gg = g_lds[2 * NHID + t], og = g_lds[3 * NHID + t];
      c = fg * c + ig * gg;
      h_lds[t] = og * tanhf(c);
    }
    __syncthreads();
    if (t < R_DIM) {                                // fused ys @ dense.T, tanh, accumulate
      float acc = db[t];
#pragma unroll
      for (int k = 0; k < NHID; ++k) acc += dw[t * NHID + k] * h_lds[k];
      out[s * R_DIM + t] += tanhf(acc);
    }
    xg = xg_next;
  }
  __syncthreads();
  if (t < NHID) { state[t] = h_lds[t]; state[NHID + t] = c; }
}

// ---------------- host orchestration ----------------
static inline int cdiv(int a, int b) { return (a + b - 1) / b; }

extern "C" void kernel_launch(void* const* d_in, const int* in_sizes, int n_in,
                              void* d_out, int out_size, void* d_ws, size_t ws_size,
                              hipStream_t stream) {
  const float*     H    = (const float*)d_in[0];
  const float*     Wm   = (const float*)d_in[1];
  const long long* HA   = (const long long*)d_in[2];
  const long long* WA   = (const long long*)d_in[3];
  const float* hg0w = (const float*)d_in[4];
  const float* hg0b = (const float*)d_in[5];
  const float* hg1w = (const float*)d_in[6];
  const float* hg1b = (const float*)d_in[7];
  const float* wg0w = (const float*)d_in[8];
  const float* wg0b = (const float*)d_in[9];
  const float* wg1w = (const float*)d_in[10];
  const float* wg1b = (const float*)d_in[11];
  const float* W_ih = (const float*)d_in[12];
  const float* W_hh = (const float*)d_in[13];
  const float* b_ih = (const float*)d_in[14];
  const float* b_hh = (const float*)d_in[15];
  const float* dHw  = (const float*)d_in[16];
  const float* dHb  = (const float*)d_in[17];
  const float* dWw  = (const float*)d_in[18];
  const float* dWb  = (const float*)d_in[19];

  float* outH = (float*)d_out;
  float* outW = outH + M_NODES * R_DIM;

  // workspace carve (≈22.5 MB of f32)
  float* p = (float*)d_ws;
  float* dinvH = p; p += M_NODES;
  float* dinvW = p; p += M_NODES;
  float* degT  = p; p += M_NODES;
  float* XW0   = p; p += M_NODES * HID0;
  float* AGG0  = p; p += M_NODES * HID0;
  float* XW1   = p; p += M_NODES * Q_DIM;
  float* AGG1  = p; p += M_NODES * Q_DIM;
  float* Htil  = p; p += M_NODES * Q_DIM;
  float* XG_H  = p; p += M_NODES * G4;
  float* XG_W  = p; p += M_NODES * G4;
  float* state = p; p += 128;

  const int BT = 256;
  const int mtiles = M_NODES / 16;                 // 625 (exact)
  const int gemmB_16  = cdiv(mtiles * (HID0 / 16), 8);   // N=16
  const int gemmB_48  = cdiv(mtiles * (Q_DIM / 16), 8);  // N=48
  const int gemmB_192 = cdiv(mtiles * (G4 / 16), 8);     // N=192
  const dim3 eG16(cdiv(NEDGE, BT), HID0 / 4);
  const dim3 eG48(cdiv(NEDGE, BT), Q_DIM / 4);

  // ---- degrees (both graphs), output init, LSTM state init ----
  zero_f32<<<cdiv(M_NODES, BT), BT, 0, stream>>>(degT, M_NODES);
  deg_accum<<<cdiv(NEDGE, BT), BT, 0, stream>>>(HA + NEDGE, degT, NEDGE);
  deg_finalize<<<cdiv(M_NODES, BT), BT, 0, stream>>>(degT, dinvH, M_NODES);
  zero_f32<<<cdiv(M_NODES, BT), BT, 0, stream>>>(degT, M_NODES);
  deg_accum<<<cdiv(NEDGE, BT), BT, 0, stream>>>(WA + NEDGE, degT, NEDGE);
  deg_finalize<<<cdiv(M_NODES, BT), BT, 0, stream>>>(degT, dinvW, M_NODES);

  copy_f32<<<cdiv(M_NODES * R_DIM, BT), BT, 0, stream>>>(H,  outH, M_NODES * R_DIM);
  copy_f32<<<cdiv(M_NODES * R_DIM, BT), BT, 0, stream>>>(Wm, outW, M_NODES * R_DIM);
  zero_f32<<<1, 128, 0, stream>>>(state, 96);

  // ---- loop-invariant H branch: Htilde + its LSTM gate precompute ----
  wmma_gemm_f16<R_DIM, HID0, false><<<gemmB_16, BT, 0, stream>>>(H, hg0w, nullptr, XW0, M_NODES);
  zero_f32<<<cdiv(M_NODES * HID0, BT), BT, 0, stream>>>(AGG0, M_NODES * HID0);
  gcn_edge_agg<HID0><<<eG16, BT, 0, stream>>>(XW0, HA, HA + NEDGE, dinvH, AGG0);
  gcn_self_bias<HID0, false><<<cdiv(M_NODES * HID0, BT), BT, 0, stream>>>(XW0, dinvH, hg0b, AGG0, AGG0, M_NODES);
  wmma_gemm_f16<HID0, Q_DIM, false><<<gemmB_48, BT, 0, stream>>>(AGG0, hg1w, nullptr, XW1, M_NODES);
  zero_f32<<<cdiv(M_NODES * Q_DIM, BT), BT, 0, stream>>>(AGG1, M_NODES * Q_DIM);
  gcn_edge_agg<Q_DIM><<<eG48, BT, 0, stream>>>(XW1, HA, HA + NEDGE, dinvH, AGG1);
  gcn_self_bias<Q_DIM, true><<<cdiv(M_NODES * Q_DIM, BT), BT, 0, stream>>>(XW1, dinvH, hg1b, AGG1, Htil, M_NODES);
  wmma_gemm_f16<Q_DIM, G4, true><<<gemmB_192, BT, 0, stream>>>(Htil, W_ih, b_ih, XG_H, M_NODES);

  // ---- T diffusion iterations ----
  for (int it = 0; it < T_STEPS; ++it) {
    lstm_dense_kernel<<<1, G4, 0, stream>>>(XG_H, W_hh, b_hh, dHw, dHb, state, outH, M_NODES);

    // W-branch GCN (consumes current Wout)
    wmma_gemm_f16<R_DIM, HID0, false><<<gemmB_16, BT, 0, stream>>>(outW, wg0w, nullptr, XW0, M_NODES);
    zero_f32<<<cdiv(M_NODES * HID0, BT), BT, 0, stream>>>(AGG0, M_NODES * HID0);
    gcn_edge_agg<HID0><<<eG16, BT, 0, stream>>>(XW0, WA, WA + NEDGE, dinvW, AGG0);
    gcn_self_bias<HID0, false><<<cdiv(M_NODES * HID0, BT), BT, 0, stream>>>(XW0, dinvW, wg0b, AGG0, AGG0, M_NODES);
    wmma_gemm_f16<HID0, Q_DIM, false><<<gemmB_48, BT, 0, stream>>>(AGG0, wg1w, nullptr, XW1, M_NODES);
    zero_f32<<<cdiv(M_NODES * Q_DIM, BT), BT, 0, stream>>>(AGG1, M_NODES * Q_DIM);
    gcn_edge_agg<Q_DIM><<<eG48, BT, 0, stream>>>(XW1, WA, WA + NEDGE, dinvW, AGG1);
    gcn_self_bias<Q_DIM, true><<<cdiv(M_NODES * Q_DIM, BT), BT, 0, stream>>>(XW1, dinvW, wg1b, AGG1, AGG1, M_NODES);
    wmma_gemm_f16<Q_DIM, G4, true><<<gemmB_192, BT, 0, stream>>>(AGG1, W_ih, b_ih, XG_W, M_NODES);

    lstm_dense_kernel<<<1, G4, 0, stream>>>(XG_W, W_hh, b_hh, dWw, dWb, state, outW, M_NODES);
  }
}
